// CustomCrossModalAttention_14826227106187
// MI455X (gfx1250) — matrix operand order repaired and checked by reading, not compiled, not run on hardware
//
#include <hip/hip_runtime.h>

// ---------- types ----------
typedef __fp16   h2raw __attribute__((ext_vector_type(2)));   // return type of cvt_pkrtz
typedef _Float16 v8h  __attribute__((ext_vector_type(8)));
typedef _Float16 v16h __attribute__((ext_vector_type(16)));
typedef float    v8f  __attribute__((ext_vector_type(8)));

struct HPair { v8h lo; v8h hi; };
struct U4 { unsigned u[4]; };
struct U8 { unsigned u[8]; };

#define DEV static __device__ __forceinline__

DEV v16h hcat(v8h a, v8h b) { HPair p; p.lo = a; p.hi = b; return __builtin_bit_cast(v16h, p); }

DEV v8f vzero() {
  v8f r;
#pragma unroll
  for (int i = 0; i < 8; ++i) r[i] = 0.0f;
  return r;
}

DEV v8f wmma16(v16h a, v16h b, v8f c) {
  // D = A(16x32 f16) * B(32x16 f16) + C(16x16 f32)
  return __builtin_amdgcn_wmma_f32_16x16x32_f16(false, a, false, b, (short)0, c, false, false);
}

DEV unsigned pkh2(float a, float b) {
  h2raw t = __builtin_amdgcn_cvt_pkrtz(a, b);
  return __builtin_bit_cast(unsigned, t);
}

// ---------- problem constants ----------
static constexpr int DIMX   = 768;
static constexpr int HEADS  = 12;
static constexpr int HD     = 64;
static constexpr int NSEQ   = 2048;
static constexpr int BATCH  = 2;
static constexpr int BN     = BATCH * NSEQ;   // 4096 rows

// ---------- workspace layout (bytes) ----------
static constexpr size_t SZ_ACT16 = (size_t)BN * DIMX * 2;       // 6291456
static constexpr size_t SZ_W16   = (size_t)DIMX * DIMX * 2;     // 1179648
static constexpr size_t SZ_GW16  = (size_t)DIMX * 2 * DIMX * 2; // 2359296
static constexpr size_t SZ_ACT32 = (size_t)BN * DIMX * 4;       // 12582912

static constexpr size_t OFF_VIS16 = 0;
static constexpr size_t OFF_IR16  = OFF_VIS16 + SZ_ACT16;
static constexpr size_t OFF_WQ16  = OFF_IR16  + SZ_ACT16;
static constexpr size_t OFF_WK16  = OFF_WQ16  + SZ_W16;
static constexpr size_t OFF_WV16  = OFF_WK16  + SZ_W16;
static constexpr size_t OFF_WO16  = OFF_WV16  + SZ_W16;
static constexpr size_t OFF_GW16  = OFF_WO16  + SZ_W16;
static constexpr size_t OFF_Q16   = OFF_GW16  + SZ_GW16;
static constexpr size_t OFF_KP16  = OFF_Q16   + SZ_ACT16;
static constexpr size_t OFF_VT16  = OFF_KP16  + SZ_ACT16;
static constexpr size_t OFF_ATT16 = OFF_VT16  + SZ_ACT16;
static constexpr size_t OFF_OUT16 = OFF_ATT16 + SZ_ACT16;
static constexpr size_t OFF_P32   = OFF_OUT16 + SZ_ACT16;
// fp32 pool: YQ/YK/YV during projections; OUT32/Z32 alias YQ/YK afterwards.

// ---------- fp32 -> f16 convert ----------
__global__ void cma_cvt_f16(const float* __restrict__ src, _Float16* __restrict__ dst, int n) {
  int i = blockIdx.x * 256 + threadIdx.x;
  if (i < n) dst[i] = (_Float16)src[i];
}

// ---------- generic WMMA GEMM: C[m,j] = sum_k A[m,k] * Bw[j,k] + bias[j] ----------
// A row = A0[k] for k < kSplit (row stride = A1 ? kSplit : K), then A1[k-kSplit].
// K must be a multiple of 64 (768 / 1536 here) -> register ping-pong double buffering.
__global__ __launch_bounds__(128)
void cma_gemm_wmma(const _Float16* __restrict__ A0,
                   const _Float16* __restrict__ A1, int kSplit,
                   const _Float16* __restrict__ Bw,
                   const float* __restrict__ bias,
                   float* __restrict__ C32,
                   _Float16* __restrict__ C16,
                   int N, int K) {
  const int wave = threadIdx.x >> 5;
  const int lane = threadIdx.x & 31;
  const int hi   = lane >> 4;
  const int ln   = lane & 15;
  const int row0 = blockIdx.x * 64 + wave * 16;
  const int col0 = blockIdx.y * 64;
  const int m    = row0 + ln;

  const int lda0 = A1 ? kSplit : K;
  const _Float16* arow0 = A0 + (size_t)m * lda0;
  const _Float16* arow1 = A1 ? (A1 + (size_t)m * (K - kSplit)) : nullptr;

  const _Float16* bp[4];
#pragma unroll
  for (int t = 0; t < 4; ++t) bp[t] = Bw + (size_t)(col0 + t * 16 + ln) * K;

  v8f acc[4];
#pragma unroll
  for (int t = 0; t < 4; ++t) acc[t] = vzero();

  // A-tile 16x32 f16: lane m = ln; k = hi*16 + e (16 contiguous halves)
  // B-tile 32x16 f16: lane n = ln; k = hi*16 + e (16 contiguous halves)
  auto loadAB = [&](v16h& a, v16h (&b)[4], int k0) {
    const _Float16* ap = (A1 && k0 >= kSplit) ? (arow1 + (k0 - kSplit)) : (arow0 + k0);
    a = *(const v16h*)(ap + hi * 16);
#pragma unroll
    for (int t = 0; t < 4; ++t) b[t] = *(const v16h*)(bp[t] + k0 + hi * 16);
  };

  v16h a0, a1, b0[4], b1[4];
  loadAB(a0, b0, 0);
  for (int k0 = 0; k0 < K; k0 += 64) {
    // loads for second half-chunk in flight while first half computes
    loadAB(a1, b1, k0 + 32);
#pragma unroll
    for (int t = 0; t < 4; ++t) acc[t] = wmma16(a0, b0[t], acc[t]);
    if (k0 + 64 < K) loadAB(a0, b0, k0 + 64);
#pragma unroll
    for (int t = 0; t < 4; ++t) acc[t] = wmma16(a1, b1[t], acc[t]);
  }

#pragma unroll
  for (int t = 0; t < 4; ++t) {
    const int col = col0 + t * 16 + ln;
    const float bv = bias ? bias[col] : 0.0f;
#pragma unroll
    for (int r = 0; r < 8; ++r) {
      const int row = row0 + r + 8 * hi;      // C layout: VGPR r -> M = r + 8*hi
      const float v = acc[t][r] + bv;
      C32[(size_t)row * N + col] = v;
      if (C16) C16[(size_t)row * N + col] = (_Float16)v;
    }
  }
}

// ---------- block reduction helper ----------
DEV void blockReduce2(float& s, float& ss) {
#pragma unroll
  for (int o = 16; o > 0; o >>= 1) {
    s  += __shfl_xor(s,  o, 32);
    ss += __shfl_xor(ss, o, 32);
  }
  __shared__ float sh0[8], sh1[8];
  const int w = threadIdx.x >> 5, l = threadIdx.x & 31;
  if (l == 0) { sh0[w] = s; sh1[w] = ss; }
  __syncthreads();
  if (threadIdx.x == 0) {
    float a = 0.f, b = 0.f;
#pragma unroll
    for (int i = 0; i < 8; ++i) { a += sh0[i]; b += sh1[i]; }
    sh0[0] = a; sh1[0] = b;
  }
  __syncthreads();
  s = sh0[0]; ss = sh1[0];
}

// ---------- LayerNorm + head repack ----------
// mode 0: Q  -> [B,H,N,64] f16
// mode 1: K' = (LN(k)*scale + pos) * log2(e) -> [B,H,N,64] f16
// mode 2: V  -> transposed [B,H,64,N] f16
__global__ __launch_bounds__(256)
void cma_ln_pack(const float* __restrict__ Y,
                 const float* __restrict__ w, const float* __restrict__ bias,
                 const float* __restrict__ pos, _Float16* __restrict__ dst, int mode) {
  const int row = blockIdx.x;
  const int bb  = row >> 11;
  const int n   = row & 2047;
  const float* y = Y + (size_t)row * DIMX;

  float x[3]; float s = 0.f, ss = 0.f;
#pragma unroll
  for (int i = 0; i < 3; ++i) {
    x[i] = y[threadIdx.x + i * 256];
    s += x[i]; ss += x[i] * x[i];
  }
  blockReduce2(s, ss);
  const float mu   = s * (1.0f / DIMX);
  const float var  = ss * (1.0f / DIMX) - mu * mu;
  const float rstd = rsqrtf(var + 1e-5f);

#pragma unroll
  for (int i = 0; i < 3; ++i) {
    const int d = threadIdx.x + i * 256;
    float val = (x[i] - mu) * rstd * w[d] + bias[d];
    const int h = d >> 6, dd = d & 63;
    if (mode == 1) val = (val * 0.125f + pos[(size_t)n * DIMX + d]) * 1.44269504f;
    size_t idx;
    if (mode == 2) idx = ((size_t)(bb * HEADS + h) * HD + dd) * NSEQ + n;
    else           idx = ((size_t)(bb * HEADS + h) * NSEQ + n) * HD + dd;
    dst[idx] = (_Float16)val;
  }
}

// ---------- flash attention, one wave = 16 query rows ----------
__global__ __launch_bounds__(128)
void cma_attn(const _Float16* __restrict__ q16, const _Float16* __restrict__ kp16,
              const _Float16* __restrict__ vT16, _Float16* __restrict__ att16) {
  const int wave = threadIdx.x >> 5, lane = threadIdx.x & 31;
  const int hi = lane >> 4, ln = lane & 15;
  const int bh = blockIdx.y;
  const int bb = bh / HEADS, h = bh % HEADS;
  const int qt = (blockIdx.x * 4 + wave) * 16;

  const _Float16* Qb = q16  + ((size_t)bh * NSEQ + qt) * HD;
  const _Float16* Kb = kp16 + (size_t)bh * NSEQ * HD;
  const _Float16* Vb = vT16 + (size_t)bh * HD * NSEQ;

  // Q^T as two B-tiles (d-slices 0..31 and 32..63); lane holds query qt+ln
  const _Float16* qrow = Qb + ln * HD;
  const v16h qb0 = hcat(*(const v8h*)(qrow + hi * 8), *(const v8h*)(qrow + 16 + hi * 8));
  const v16h qb1 = hcat(*(const v8h*)(qrow + 32 + hi * 8), *(const v8h*)(qrow + 48 + hi * 8));

  v8f o[4];
#pragma unroll
  for (int t = 0; t < 4; ++t) o[t] = vzero();
  float mI = -1e30f, lI = 0.0f;

  for (int kc = 0; kc < NSEQ; kc += 32) {
    // --- issue ALL loads for this chunk first (K' tiles + V^T tiles) ---
    const _Float16* kr0 = Kb + (size_t)(kc + ln) * HD;
    const _Float16* kr1 = kr0 + (size_t)16 * HD;
    v16h ka[4];
    ka[0] = hcat(*(const v8h*)(kr0 + hi * 8),      *(const v8h*)(kr0 + 16 + hi * 8));
    ka[1] = hcat(*(const v8h*)(kr0 + 32 + hi * 8), *(const v8h*)(kr0 + 48 + hi * 8));
    ka[2] = hcat(*(const v8h*)(kr1 + hi * 8),      *(const v8h*)(kr1 + 16 + hi * 8));
    ka[3] = hcat(*(const v8h*)(kr1 + 32 + hi * 8), *(const v8h*)(kr1 + 48 + hi * 8));
    v16h va[4];
#pragma unroll
    for (int t = 0; t < 4; ++t) {
      const _Float16* vr = Vb + (size_t)(t * 16 + ln) * NSEQ + kc;
      va[t] = *(const v16h*)(vr + hi * 16);
    }

    // --- S^T tiles: T[key, q] = K'[key] . Q[q] (scale + pos + log2e folded into K') ---
    v8f t0 = vzero(), t1 = vzero();
    t0 = wmma16(ka[0], qb0, t0);
    t0 = wmma16(ka[1], qb1, t0);
    t1 = wmma16(ka[2], qb0, t1);
    t1 = wmma16(ka[3], qb1, t1);

    // --- online softmax; query = ln is the lane dim, keys in-lane + xor16 ---
    float cm = -1e30f;
#pragma unroll
    for (int r = 0; r < 8; ++r) cm = fmaxf(cm, fmaxf(t0[r], t1[r]));
    cm = fmaxf(cm, __shfl_xor(cm, 16, 32));
    const float mN = fmaxf(mI, cm);
    const float al = exp2f(mI - mN);
    float p0[8], p1[8]; float s = 0.0f;
#pragma unroll
    for (int r = 0; r < 8; ++r) {
      p0[r] = exp2f(t0[r] - mN);
      p1[r] = exp2f(t1[r] - mN);
      s += p0[r] + p1[r];
    }
    s += __shfl_xor(s, 16, 32);
    lI = lI * al + s;
    mI = mN;
#pragma unroll
    for (int t = 0; t < 4; ++t)
#pragma unroll
      for (int r = 0; r < 8; ++r) o[t][r] *= al;

    // --- build P^T B-tile (32 keys x 16 queries) via half-wave swaps ---
    unsigned pk0[4], pk1[4];
#pragma unroll
    for (int j = 0; j < 4; ++j) {
      pk0[j] = pkh2(p0[2 * j], p0[2 * j + 1]);
      pk1[j] = pkh2(p1[2 * j], p1[2 * j + 1]);
    }
    U8 br;
#pragma unroll
    for (int j = 0; j < 4; ++j) {
      const unsigned x0 = (unsigned)__shfl_xor((int)pk0[j], 16, 32);
      const unsigned x1 = (unsigned)__shfl_xor((int)pk1[j], 16, 32);
      br.u[j]     = hi ? x1 : pk0[j];   // K = hi*16 + (0..7)
      br.u[4 + j] = hi ? pk1[j] : x0;   // K = hi*16 + (8..15)
    }
    const v16h bP = __builtin_bit_cast(v16h, br);

    // --- O^T += V^T(16d x 32k) * P^T(32k x 16q), 4 d-tiles ---
#pragma unroll
    for (int t = 0; t < 4; ++t) o[t] = wmma16(va[t], bP, o[t]);
  }

  const float inv = 1.0f / lI;
  const int rowTok = bb * NSEQ + qt + ln;
  _Float16* orow = att16 + (size_t)rowTok * DIMX + h * HD;
#pragma unroll
  for (int t = 0; t < 4; ++t) {
    U4 st;
#pragma unroll
    for (int j = 0; j < 4; ++j) st.u[j] = pkh2(o[t][2 * j] * inv, o[t][2 * j + 1] * inv);
    *(v8h*)(orow + t * 16 + hi * 8) = __builtin_bit_cast(v8h, st);  // d = t*16 + r + 8*hi
  }
}

// ---------- sigmoid gate fuse + final LayerNorm ----------
__global__ __launch_bounds__(256)
void cma_gate_fuse_ln(const float* __restrict__ vis, const float* __restrict__ out32,
                      const float* __restrict__ z32, const float* __restrict__ lnw,
                      const float* __restrict__ lnb, float* __restrict__ dout) {
  const int row = blockIdx.x;
  const size_t base = (size_t)row * DIMX;
  float f[3]; float s = 0.f, ss = 0.f;
#pragma unroll
  for (int i = 0; i < 3; ++i) {
    const int d = threadIdx.x + i * 256;
    const float z = z32[base + d];
    const float g = 1.0f / (1.0f + __expf(-z));
    const float v = vis[base + d];
    const float o = out32[base + d];
    const float fv = v * g + o * (1.0f - g);
    f[i] = fv; s += fv; ss += fv * fv;
  }
  blockReduce2(s, ss);
  const float mu   = s * (1.0f / DIMX);
  const float var  = ss * (1.0f / DIMX) - mu * mu;
  const float rstd = rsqrtf(var + 1e-5f);
#pragma unroll
  for (int i = 0; i < 3; ++i) {
    const int d = threadIdx.x + i * 256;
    dout[base + d] = (f[i] - mu) * rstd * lnw[d] + lnb[d];
  }
}

// ---------- launcher ----------
extern "C" void kernel_launch(void* const* d_in, const int* in_sizes, int n_in,
                              void* d_out, int out_size, void* d_ws, size_t ws_size,
                              hipStream_t stream) {
  const float* vis   = (const float*)d_in[0];
  const float* ir    = (const float*)d_in[1];
  const float* wq    = (const float*)d_in[2];
  const float* bq    = (const float*)d_in[3];
  const float* lnq_w = (const float*)d_in[4];
  const float* lnq_b = (const float*)d_in[5];
  const float* wk    = (const float*)d_in[6];
  const float* bk    = (const float*)d_in[7];
  const float* lnk_w = (const float*)d_in[8];
  const float* lnk_b = (const float*)d_in[9];
  const float* wv    = (const float*)d_in[10];
  const float* bv    = (const float*)d_in[11];
  const float* lnv_w = (const float*)d_in[12];
  const float* lnv_b = (const float*)d_in[13];
  const float* pos   = (const float*)d_in[14];
  const float* wo    = (const float*)d_in[15];
  const float* bo    = (const float*)d_in[16];
  const float* gw    = (const float*)d_in[17];
  const float* gb    = (const float*)d_in[18];
  const float* lnw   = (const float*)d_in[19];
  const float* lnb   = (const float*)d_in[20];
  float* dout = (float*)d_out;

  char* ws = (char*)d_ws;
  _Float16* vis16 = (_Float16*)(ws + OFF_VIS16);
  _Float16* ir16  = (_Float16*)(ws + OFF_IR16);
  _Float16* wq16  = (_Float16*)(ws + OFF_WQ16);
  _Float16* wk16  = (_Float16*)(ws + OFF_WK16);
  _Float16* wv16  = (_Float16*)(ws + OFF_WV16);
  _Float16* wo16  = (_Float16*)(ws + OFF_WO16);
  _Float16* gw16  = (_Float16*)(ws + OFF_GW16);
  _Float16* q16   = (_Float16*)(ws + OFF_Q16);
  _Float16* kp16  = (_Float16*)(ws + OFF_KP16);
  _Float16* vT16  = (_Float16*)(ws + OFF_VT16);
  _Float16* att16 = (_Float16*)(ws + OFF_ATT16);
  _Float16* out16 = (_Float16*)(ws + OFF_OUT16);
  float* yq32  = (float*)(ws + OFF_P32);
  float* yk32  = (float*)(ws + OFF_P32 + SZ_ACT32);
  float* yv32  = (float*)(ws + OFF_P32 + 2 * SZ_ACT32);
  float* out32 = yq32;  // alias: Y buffers dead after ln_pack
  float* z32   = yk32;

  // 1) converts
  const int nAct = BN * DIMX, nW = DIMX * DIMX, nGW = DIMX * 2 * DIMX;
  cma_cvt_f16<<<(nAct + 255) / 256, 256, 0, stream>>>(vis, vis16, nAct);
  cma_cvt_f16<<<(nAct + 255) / 256, 256, 0, stream>>>(ir, ir16, nAct);
  cma_cvt_f16<<<(nW + 255) / 256, 256, 0, stream>>>(wq, wq16, nW);
  cma_cvt_f16<<<(nW + 255) / 256, 256, 0, stream>>>(wk, wk16, nW);
  cma_cvt_f16<<<(nW + 255) / 256, 256, 0, stream>>>(wv, wv16, nW);
  cma_cvt_f16<<<(nW + 255) / 256, 256, 0, stream>>>(wo, wo16, nW);
  cma_cvt_f16<<<(nGW + 255) / 256, 256, 0, stream>>>(gw, gw16, nGW);

  // 2) QKV projections
  dim3 gb128(128);
  dim3 ggrid(BN / 64, DIMX / 64);
  cma_gemm_wmma<<<ggrid, gb128, 0, stream>>>(ir16,  nullptr, DIMX, wq16, bq, yq32, nullptr, DIMX, DIMX);
  cma_gemm_wmma<<<ggrid, gb128, 0, stream>>>(vis16, nullptr, DIMX, wk16, bk, yk32, nullptr, DIMX, DIMX);
  cma_gemm_wmma<<<ggrid, gb128, 0, stream>>>(vis16, nullptr, DIMX, wv16, bv, yv32, nullptr, DIMX, DIMX);

  // 3) LN + repack (K' folds scale, positional term and log2(e))
  cma_ln_pack<<<BN, 256, 0, stream>>>(yq32, lnq_w, lnq_b, pos, q16, 0);
  cma_ln_pack<<<BN, 256, 0, stream>>>(yk32, lnk_w, lnk_b, pos, kp16, 1);
  cma_ln_pack<<<BN, 256, 0, stream>>>(yv32, lnv_w, lnv_b, pos, vT16, 2);

  // 4) attention
  cma_attn<<<dim3(NSEQ / 16 / 4, BATCH * HEADS), 128, 0, stream>>>(q16, kp16, vT16, att16);

  // 5) output projection (fp32 for epilogue + f16 for gate GEMM)
  cma_gemm_wmma<<<ggrid, gb128, 0, stream>>>(att16, nullptr, DIMX, wo16, bo, out32, out16, DIMX, DIMX);

  // 6) gate GEMM over concat(visible, out), K = 1536 split across two A pointers
  cma_gemm_wmma<<<ggrid, gb128, 0, stream>>>(vis16, out16, DIMX, gw16, gb, z32, nullptr, DIMX, 2 * DIMX);

  // 7) fuse + final LN
  cma_gate_fuse_ln<<<BN, 256, 0, stream>>>(vis, out32, z32, lnw, lnb, dout);
}
